// PaiNN_76940044140995
// MI455X (gfx1250) — matrix operand order repaired and checked by reading, not compiled, not run on hardware
//
#include <hip/hip_runtime.h>
#include <hip/hip_bf16.h>
#include <math.h>

#define DD   128
#define NB   20
#define LL   3
#define HID  384          // 3*DD
#define CUTR 5.0f
#define PI_F 3.14159265358979323846f

typedef __bf16 v16bf __attribute__((ext_vector_type(16)));
typedef float  v8f   __attribute__((ext_vector_type(8)));

__device__ __forceinline__ __bf16 f2bf(float f) {
    __hip_bfloat16 h = __float2bfloat16(f);
    __bf16 o;
    __builtin_memcpy(&o, &h, 2);
    return o;
}

__device__ __forceinline__ float silu_f(float v) {
    return v * (1.0f / (1.0f + __expf(-v)));
}

// ---------------------------------------------------------------------------
// Tiled WMMA GEMM:  C[M,Nout] = act(A[M,K] @ W[K,Nout] + bias)
//   A: f32 row-major [M,K] (converted to bf16 in registers)
//   Wt: bf16 col-major [Nout,K] (pre-transposed weights)
// One wave computes a 16x32 output tile (two 16x16 WMMA tiles sharing the
// A fragment). A and B fragments both use contiguous K order, k = 16*(lane/16) + s (wave32).
// ---------------------------------------------------------------------------
template <bool SILU, int K>
__global__ __launch_bounds__(256)
void gemm_wmma(const float* __restrict__ A, const __bf16* __restrict__ Wt,
               const float* __restrict__ bias, float* __restrict__ C,
               int M, int Nout) {
    const int lane  = threadIdx.x & 31;
    const int wave  = threadIdx.x >> 5;
    const int wpb   = blockDim.x >> 5;
    const int row0  = blockIdx.x * 16;
    const int col0  = (blockIdx.y * wpb + wave) * 32;   // 32 cols per wave

    const int  m    = lane & 15;
    int row = row0 + m;
    row = (row < M) ? row : (M - 1);                    // clamp: no predication
    const int abase = (lane < 16) ? 0 : 16;             // A K-base (this half)
    const int bbase = (lane < 16) ? 0 : 16;             // B K-base (this half)
    const int bcol0 = col0 + (lane & 15);
    const int bcol1 = bcol0 + 16;

    const float*  arow = A + (long long)row * K;
    const __bf16* wc0  = Wt + (size_t)bcol0 * K;
    const __bf16* wc1  = Wt + (size_t)bcol1 * K;

    v8f acc0 = {}, acc1 = {};
#pragma unroll
    for (int k0 = 0; k0 < K; k0 += 32) {
        if (k0 + 32 < K)
            __builtin_prefetch(arow + k0 + 32, 0, 3);   // -> global_prefetch_b8

        // A fragment: K = k0+abase+0..15 (contiguous 4-vecs)
        const float4 x0 = *(const float4*)(arow + k0 + abase);
        const float4 x1 = *(const float4*)(arow + k0 + abase + 4);
        const float4 x2 = *(const float4*)(arow + k0 + abase + 8);
        const float4 x3 = *(const float4*)(arow + k0 + abase + 12);
        v16bf a;
        a[0]  = f2bf(x0.x); a[1]  = f2bf(x0.y); a[2]  = f2bf(x0.z); a[3]  = f2bf(x0.w);
        a[4]  = f2bf(x1.x); a[5]  = f2bf(x1.y); a[6]  = f2bf(x1.z); a[7]  = f2bf(x1.w);
        a[8]  = f2bf(x2.x); a[9]  = f2bf(x2.y); a[10] = f2bf(x2.z); a[11] = f2bf(x2.w);
        a[12] = f2bf(x3.x); a[13] = f2bf(x3.y); a[14] = f2bf(x3.z); a[15] = f2bf(x3.w);

        // B fragments: contiguous 16 bf16 along K (32-byte vector loads)
        const v16bf b0 = *(const v16bf*)(wc0 + k0 + bbase);
        const v16bf b1 = *(const v16bf*)(wc1 + k0 + bbase);

        acc0 = __builtin_amdgcn_wmma_f32_16x16x32_bf16(
                   false, a, false, b0, (short)0, acc0, false, false);
        acc1 = __builtin_amdgcn_wmma_f32_16x16x32_bf16(
                   false, a, false, b1, (short)0, acc1, false, false);
    }

    const int   mbase = (lane < 16) ? 0 : 8;
    const float bv0   = bias ? bias[bcol0] : 0.0f;
    const float bv1   = bias ? bias[bcol1] : 0.0f;
#pragma unroll
    for (int j = 0; j < 8; ++j) {
        const int r = row0 + mbase + j;
        if (r < M) {
            float v0 = acc0[j] + bv0;
            float v1 = acc1[j] + bv1;
            if (SILU) { v0 = silu_f(v0); v1 = silu_f(v1); }
            C[(long long)r * Nout + bcol0] = v0;
            C[(long long)r * Nout + bcol1] = v1;
        }
    }
}

// Pre-transpose weights: W f32 [K,Nout] -> Wt bf16 [Nout,K]
__global__ void wt_bf16(const float* __restrict__ W, __bf16* __restrict__ Wt,
                        int K, int Nout) {
    int i = blockIdx.x * blockDim.x + threadIdx.x;
    if (i >= K * Nout) return;
    int k = i / Nout, n = i % Nout;
    Wt[(size_t)n * K + k] = f2bf(W[i]);
}

// ---------------------------------------------------------------------------
// Elementwise / edge kernels
// ---------------------------------------------------------------------------
__global__ void zero_kernel(float* __restrict__ p, long long n) {
    long long i = (long long)blockIdx.x * blockDim.x + threadIdx.x;
    if (i < n) p[i] = 0.0f;
}

__global__ void add_inplace(float* __restrict__ p, const float* __restrict__ q,
                            long long n) {
    long long i = (long long)blockIdx.x * blockDim.x + threadIdx.x;
    if (i < n) p[i] += q[i];
}

__global__ void init_xs(const int* __restrict__ at_no,
                        const float* __restrict__ embed_w,
                        float* __restrict__ xs, int N) {
    long long t = (long long)blockIdx.x * blockDim.x + threadIdx.x;
    if (t >= (long long)N * DD) return;
    int n = (int)(t >> 7), d = (int)(t & 127);
    xs[t] = embed_w[at_no[n] * DD + d];
}

__global__ void edge_geom(const float* __restrict__ pos,
                          const int* __restrict__ ei,
                          float* __restrict__ rsh, float* __restrict__ rbf,
                          float* __restrict__ fcut, int E) {
    int e = blockIdx.x * blockDim.x + threadIdx.x;
    if (e >= E) return;
    int s = ei[e], d = ei[E + e];
    float vx = pos[d * 3 + 0] - pos[s * 3 + 0];
    float vy = pos[d * 3 + 1] - pos[s * 3 + 1];
    float vz = pos[d * 3 + 2] - pos[s * 3 + 2];
    float dist = sqrtf(vx * vx + vy * vy + vz * vz);
    float inv  = 1.0f / dist;
    rsh[e * 3 + 0] = vx * inv;
    rsh[e * 3 + 1] = vy * inv;
    rsh[e * 3 + 2] = vz * inv;
    float pref = sqrtf(2.0f / CUTR) * inv;
    float arg  = PI_F * dist / CUTR;
#pragma unroll
    for (int k = 0; k < NB; ++k)
        rbf[(long long)e * NB + k] = pref * sinf((float)(k + 1) * arg);
    fcut[e] = (dist < CUTR) ? 0.5f * (cosf(arg) + 1.0f) : 0.0f;
}

// fw = (rbf @ rbf_w + rbf_b) * fcut ; fo = h[dst]*fw ; scatter messages.
// One thread per (edge, feature d). m_s -> xs (atomic), m_v -> dxv (atomic).
__global__ void edge_msg(const float* __restrict__ h,
                         const float* __restrict__ rbf,
                         const float* __restrict__ fcut,
                         const float* __restrict__ rsh,
                         const int* __restrict__ ei,
                         const float* __restrict__ rbf_w,  // [NB,HID] layer slice
                         const float* __restrict__ rbf_b,  // [HID]
                         const float* __restrict__ xv,
                         float* __restrict__ xs, float* __restrict__ dxv,
                         int E) {
    long long t = (long long)blockIdx.x * blockDim.x + threadIdx.x;
    if (t >= (long long)E * DD) return;
    int e = (int)(t >> 7), d = (int)(t & 127);
    int s = ei[e], dst = ei[E + e];

    float f0 = rbf_b[d], f1 = rbf_b[DD + d], f2 = rbf_b[2 * DD + d];
    const float* r = rbf + (long long)e * NB;
#pragma unroll
    for (int k = 0; k < NB; ++k) {
        float rv = r[k];
        f0 += rv * rbf_w[k * HID + d];
        f1 += rv * rbf_w[k * HID + DD + d];
        f2 += rv * rbf_w[k * HID + 2 * DD + d];
    }
    float fc = fcut[e];
    const float* hd = h + (long long)dst * HID;
    float ms = hd[d]          * f0 * fc;   // m_s
    float ge = hd[DD + d]     * f1 * fc;   // g_edge
    float gv = hd[2 * DD + d] * f2 * fc;   // g_vec

    atomicAdd(&xs[(long long)s * DD + d], ms);

    float rx = rsh[e * 3 + 0], ry = rsh[e * 3 + 1], rz = rsh[e * 3 + 2];
    const float* xvd = xv + (long long)dst * 3 * DD;
    float* dvs = dxv + (long long)s * 3 * DD;
    atomicAdd(&dvs[0 * DD + d], xvd[0 * DD + d] * gv + ge * rx);
    atomicAdd(&dvs[1 * DD + d], xvd[1 * DD + d] * gv + ge * ry);
    atomicAdd(&dvs[2 * DD + d], xvd[2 * DD + d] * gv + ge * rz);
}

// cat = [x_scalar, ||V||_channels]  -> [N, 2D]
__global__ void build_cat(const float* __restrict__ xs,
                          const float* __restrict__ V,
                          float* __restrict__ cat, int N) {
    long long t = (long long)blockIdx.x * blockDim.x + threadIdx.x;
    if (t >= (long long)N * DD) return;
    int n = (int)(t >> 7), d = (int)(t & 127);
    const float* vn = V + (long long)n * 3 * DD;
    float v0 = vn[d], v1 = vn[DD + d], v2 = vn[2 * DD + d];
    cat[(long long)n * 2 * DD + d]      = xs[(long long)n * DD + d];
    cat[(long long)n * 2 * DD + DD + d] = sqrtf(v0 * v0 + v1 * v1 + v2 * v2);
}

// x_vector += a_vv*U ; x_scalar += a_sv*sum_c(U*V) + a_ss
__global__ void upd_final(const float* __restrict__ mo,
                          const float* __restrict__ U,
                          const float* __restrict__ V,
                          float* __restrict__ xs, float* __restrict__ xv, int N) {
    long long t = (long long)blockIdx.x * blockDim.x + threadIdx.x;
    if (t >= (long long)N * DD) return;
    int n = (int)(t >> 7), d = (int)(t & 127);
    const float* mon = mo + (long long)n * HID;
    float ass = mon[d], avv = mon[DD + d], asv = mon[2 * DD + d];
    float dot = 0.0f;
#pragma unroll
    for (int c = 0; c < 3; ++c) {
        long long idx = ((long long)n * 3 + c) * DD + d;
        float u = U[idx];
        dot += u * V[idx];
        xv[idx] += avv * u;
    }
    xs[(long long)n * DD + d] += asv * dot + ass;
}

__global__ void readout(const float* __restrict__ tr,
                        const float* __restrict__ w2,
                        const float* __restrict__ b2,
                        const int* __restrict__ batch,
                        float* __restrict__ out, int N) {
    int n = blockIdx.x * blockDim.x + threadIdx.x;
    if (n >= N) return;
    float acc = b2[0];
    const float* tn = tr + (long long)n * 64;
#pragma unroll
    for (int k = 0; k < 64; ++k) acc += tn[k] * w2[k];
    atomicAdd(&out[batch[n]], acc);
}

// ---------------------------------------------------------------------------
static inline void launch_gemm(const float* A, const __bf16* Wt, const float* b,
                               float* C, int M, int Nout, int K, bool silu,
                               hipStream_t s) {
    int wpb = (Nout % 128 == 0) ? 4 : 2;          // each wave covers 32 cols
    dim3 block(32 * wpb);
    dim3 grid((M + 15) / 16, Nout / (32 * wpb));
    if (K == 128) {
        if (silu) gemm_wmma<true , 128><<<grid, block, 0, s>>>(A, Wt, b, C, M, Nout);
        else      gemm_wmma<false, 128><<<grid, block, 0, s>>>(A, Wt, b, C, M, Nout);
    } else {      // K == 256, only used with silu
        gemm_wmma<true, 256><<<grid, block, 0, s>>>(A, Wt, b, C, M, Nout);
    }
}

static inline dim3 g1(long long n, int bs) { return dim3((unsigned)((n + bs - 1) / bs)); }

static inline void conv_wt(const float* W, __bf16* Wt, int K, int Nout,
                           hipStream_t s) {
    wt_bf16<<<g1((long long)K * Nout, 256), 256, 0, s>>>(W, Wt, K, Nout);
}

extern "C" void kernel_launch(void* const* d_in, const int* in_sizes, int n_in,
                              void* d_out, int out_size, void* d_ws, size_t ws_size,
                              hipStream_t stream) {
    const int N = in_sizes[0];
    const int E = in_sizes[2] / 2;

    const int*   at_no   = (const int*)  d_in[0];
    const float* pos     = (const float*)d_in[1];
    const int*   ei      = (const int*)  d_in[2];
    const int*   batch   = (const int*)  d_in[3];
    const float* embed_w = (const float*)d_in[4];
    const float* msg_w1  = (const float*)d_in[5];
    const float* msg_b1  = (const float*)d_in[6];
    const float* msg_w2  = (const float*)d_in[7];
    const float* msg_b2  = (const float*)d_in[8];
    const float* rbf_w   = (const float*)d_in[9];
    const float* rbf_b   = (const float*)d_in[10];
    const float* upd_U   = (const float*)d_in[11];
    const float* upd_V   = (const float*)d_in[12];
    const float* upd_w1  = (const float*)d_in[13];
    const float* upd_b1  = (const float*)d_in[14];
    const float* upd_w2  = (const float*)d_in[15];
    const float* upd_b2  = (const float*)d_in[16];
    const float* out_w1  = (const float*)d_in[17];
    const float* out_b1  = (const float*)d_in[18];
    const float* out_w2  = (const float*)d_in[19];
    const float* out_b2  = (const float*)d_in[20];
    float*       out     = (float*)d_out;

    // workspace carve-up (f32 region)
    float* ws = (float*)d_ws;
    size_t o = 0;
    float* rsh  = ws + o; o += (size_t)E * 3;
    float* rbf  = ws + o; o += (size_t)E * NB;
    float* fcut = ws + o; o += (size_t)E;
    float* xs   = ws + o; o += (size_t)N * DD;
    float* xv   = ws + o; o += (size_t)N * 3 * DD;
    float* h    = ws + o; o += (size_t)N * HID;     // also reused as `mo`
    float* t1   = ws + o; o += (size_t)N * DD;      // MLP temp / readout temp
    float* U    = ws + o; o += (size_t)N * 3 * DD;  // also reused as dxv scratch
    float* V    = ws + o; o += (size_t)N * 3 * DD;
    float* cat  = ws + o; o += (size_t)N * 2 * DD;
    o = (o + 7) & ~(size_t)7;                       // 32B-align bf16 region

    // bf16 transposed-weight region
    __bf16* bws = (__bf16*)(ws + o);
    size_t bo = 0;
    __bf16 *wt_m1[LL], *wt_m2[LL], *wt_uU[LL], *wt_uV[LL], *wt_u1[LL], *wt_u2[LL];
    for (int l = 0; l < LL; ++l) {
        wt_m1[l] = bws + bo; bo += (size_t)DD * DD;
        wt_m2[l] = bws + bo; bo += (size_t)DD * HID;
        wt_uU[l] = bws + bo; bo += (size_t)DD * DD;
        wt_uV[l] = bws + bo; bo += (size_t)DD * DD;
        wt_u1[l] = bws + bo; bo += (size_t)2 * DD * DD;
        wt_u2[l] = bws + bo; bo += (size_t)DD * HID;
    }
    __bf16* wt_o1 = bws + bo; bo += (size_t)DD * (DD / 2);
    (void)ws_size; (void)n_in;

    const long long nxv = (long long)N * 3 * DD;

    // ---- weight pre-transpose (f32 [K,Nout] -> bf16 [Nout,K]) ----
    for (int l = 0; l < LL; ++l) {
        conv_wt(msg_w1 + (size_t)l * DD * DD,      wt_m1[l], DD,     DD,  stream);
        conv_wt(msg_w2 + (size_t)l * DD * HID,     wt_m2[l], DD,     HID, stream);
        conv_wt(upd_U  + (size_t)l * DD * DD,      wt_uU[l], DD,     DD,  stream);
        conv_wt(upd_V  + (size_t)l * DD * DD,      wt_uV[l], DD,     DD,  stream);
        conv_wt(upd_w1 + (size_t)l * 2 * DD * DD,  wt_u1[l], 2 * DD, DD,  stream);
        conv_wt(upd_w2 + (size_t)l * DD * HID,     wt_u2[l], DD,     HID, stream);
    }
    conv_wt(out_w1, wt_o1, DD, DD / 2, stream);

    // ---- setup ----
    init_xs    <<<g1((long long)N * DD, 256), 256, 0, stream>>>(at_no, embed_w, xs, N);
    zero_kernel<<<g1(nxv, 256), 256, 0, stream>>>(xv, nxv);
    edge_geom  <<<g1(E, 256), 256, 0, stream>>>(pos, ei, rsh, rbf, fcut, E);
    zero_kernel<<<g1(out_size, 256), 256, 0, stream>>>(out, out_size);

    // ---- layers ----
    for (int l = 0; l < LL; ++l) {
        const float* mb1 = msg_b1 + (size_t)l * DD;
        const float* mb2 = msg_b2 + (size_t)l * HID;
        const float* rw  = rbf_w  + (size_t)l * NB * HID;
        const float* rb  = rbf_b  + (size_t)l * HID;
        const float* ub1 = upd_b1 + (size_t)l * DD;
        const float* ub2 = upd_b2 + (size_t)l * HID;

        // message: h = silu(xs@W1+b1)@W2+b2
        launch_gemm(xs, wt_m1[l], mb1, t1, N, DD, DD, true, stream);
        launch_gemm(t1, wt_m2[l], mb2, h, N, HID, DD, false, stream);

        // edge messages -> xs (atomic) and dxv (U buffer, atomic)
        zero_kernel<<<g1(nxv, 256), 256, 0, stream>>>(U, nxv);
        edge_msg<<<g1((long long)E * DD, 256), 256, 0, stream>>>(
            h, rbf, fcut, rsh, ei, rw, rb, xv, xs, U, E);
        add_inplace<<<g1(nxv, 256), 256, 0, stream>>>(xv, U, nxv);

        // update: U/V channel transforms as [3N,D]@[D,D] WMMA GEMMs
        launch_gemm(xv, wt_uU[l], nullptr, U, 3 * N, DD, DD, false, stream);
        launch_gemm(xv, wt_uV[l], nullptr, V, 3 * N, DD, DD, false, stream);
        build_cat<<<g1((long long)N * DD, 256), 256, 0, stream>>>(xs, V, cat, N);
        launch_gemm(cat, wt_u1[l], ub1, t1, N, DD, 2 * DD, true, stream);
        launch_gemm(t1, wt_u2[l], ub2, h /*mo*/, N, HID, DD, false, stream);
        upd_final<<<g1((long long)N * DD, 256), 256, 0, stream>>>(h, U, V, xs, xv, N);
    }

    // ---- readout ----
    launch_gemm(xs, wt_o1, out_b1, t1, N, DD / 2, DD, true, stream);
    readout<<<g1(N, 256), 256, 0, stream>>>(t1, out_w2, out_b2, batch, out, N);
}
